// self_a_22574348107986
// MI455X (gfx1250) — matrix operand order repaired and checked by reading, not compiled, not run on hardware
//
#include <hip/hip_runtime.h>

// Problem constants (match the reference).
#define NB 2        // batch
#define CC 64       // channels
#define NN 9216     // pixels = 96*96
#define KB 64       // keys per inner iteration
#define KHALF (NN / 2)   // keys per wave (split-K factor 2)

typedef __attribute__((ext_vector_type(16))) __bf16 v16bf;
typedef __attribute__((ext_vector_type(8)))  __bf16 v8bf;
typedef __attribute__((ext_vector_type(8)))  float  v8f;

__device__ __forceinline__ __bf16 f2bf(float f) { return (__bf16)f; }

__device__ __forceinline__ v16bf cat16(v8bf lo, v8bf hi) {
    return __builtin_shufflevector(lo, hi, 0,1,2,3,4,5,6,7,8,9,10,11,12,13,14,15);
}

// A-fragment (16x32 bf16, row-major source, row = this lane's M).
// Contiguous K order: lanes 0-15 -> K 0..15, lanes 16-31 -> K 16..31.
__device__ __forceinline__ v16bf load_a_row(const __bf16* row, int h, int c0) {
    v8bf lo = *(const v8bf*)(row + c0 + h * 16);
    v8bf hi = *(const v8bf*)(row + c0 + h * 16 + 8);
    return cat16(lo, hi);
}

__device__ __forceinline__ v8f wmma_bf16(v16bf a, v16bf b, v8f c) {
    return __builtin_amdgcn_wmma_f32_16x16x32_bf16(
        /*neg_a=*/false, a, /*neg_b=*/false, b,
        /*c_mod=*/(short)0, c, /*reuse_a=*/false, /*reuse_b=*/false);
}

// DPP16 mirror-butterfly reduction over each 16-lane row (pure VALU, no DS):
// stages xor1 (quad_perm 0xB1), xor2 (quad_perm 0x4E), row_half_mirror (0x141),
// row_mirror (0x140). After 4 stages every lane holds the full 16-lane result.
template <int CTRL>
__device__ __forceinline__ float dpp_perm(float v) {
    int r = __builtin_amdgcn_update_dpp(0, __builtin_bit_cast(int, v),
                                        CTRL, 0xF, 0xF, true);
    return __builtin_bit_cast(float, r);
}
__device__ __forceinline__ float rowmax16(float v) {
    v = fmaxf(v, dpp_perm<0xB1>(v));
    v = fmaxf(v, dpp_perm<0x4E>(v));
    v = fmaxf(v, dpp_perm<0x141>(v));
    v = fmaxf(v, dpp_perm<0x140>(v));
    return v;
}
__device__ __forceinline__ float rowsum16(float v) {
    v += dpp_perm<0xB1>(v);
    v += dpp_perm<0x4E>(v);
    v += dpp_perm<0x141>(v);
    v += dpp_perm<0x140>(v);
    return v;
}

// ---------------------------------------------------------------------------
// Kernel 1: the four 1x1-conv projections, fp32 math, bf16 outputs.
//   Qt, Kt : (B, N, C) row-major  (query/key rows contiguous in channels)
//   Vx, Vy : (B, C, N) row-major  (channel rows contiguous in pixels)
// ---------------------------------------------------------------------------
__global__ __launch_bounds__(256) void proj_kernel(
    const float* __restrict__ x,  const float* __restrict__ y,
    const float* __restrict__ Wq, const float* __restrict__ bq,
    const float* __restrict__ Wk, const float* __restrict__ bk,
    const float* __restrict__ Wvx,const float* __restrict__ bvx,
    const float* __restrict__ Wvy,const float* __restrict__ bvy,
    __bf16* __restrict__ Qt, __bf16* __restrict__ Kt,
    __bf16* __restrict__ Vx, __bf16* __restrict__ Vy)
{
    const int z = blockIdx.z;      // 0:Q 1:K 2:Vx 3:Vy
    const int b = blockIdx.y;

    const float* in;  const float* Wm; const float* bias;
    if (z == 0)      { in = y; Wm = Wq;  bias = bq;  }
    else if (z == 1) { in = x; Wm = Wk;  bias = bk;  }
    else if (z == 2) { in = x; Wm = Wvx; bias = bvx; }
    else             { in = y; Wm = Wvy; bias = bvy; }

    int o, p;
    if (z < 2) {
        // (p broadcast, o contiguous) -> coalesced (N,C) stores
        o = threadIdx.x & 63;
        p = blockIdx.x * 4 + (threadIdx.x >> 6);
    } else {
        // (o broadcast, p contiguous) -> coalesced (C,N) stores + loads
        int pl = threadIdx.x & 63;
        int og = threadIdx.x >> 6;
        p = (blockIdx.x % (NN / 64)) * 64 + pl;
        o = (blockIdx.x / (NN / 64)) * 4 + og;
    }

    const float* ip = in + (size_t)b * CC * NN + p;
    const float* wr = Wm + (size_t)o * CC;
    float s = bias[o];
#pragma unroll 8
    for (int c = 0; c < CC; ++c) s += wr[c] * ip[(size_t)c * NN];

    __bf16 v = f2bf(s);
    if (z == 0)      Qt[((size_t)b * NN + p) * CC + o] = v;
    else if (z == 1) Kt[((size_t)b * NN + p) * CC + o] = v;
    else if (z == 2) Vx[((size_t)b * CC + o) * NN + p] = v;
    else             Vy[((size_t)b * CC + o) * NN + p] = v;
}

// ---------------------------------------------------------------------------
// Kernel 2: fused flash-style cross attention, split-K over keys.
// 128 threads = 4 waves = 2 query tiles x 2 key-halves.
// Per 64-key block and wave: 8 score WMMAs + online softmax + 16 PV WMMAs.
// The two key-half states per query tile are merged once in the epilogue.
// ---------------------------------------------------------------------------
__global__ __launch_bounds__(128) void attn_kernel(
    const __bf16* __restrict__ Qt, const __bf16* __restrict__ Kt,
    const __bf16* __restrict__ Vx, const __bf16* __restrict__ Vy,
    const float* __restrict__ gptr, const float* __restrict__ bptr,
    float* __restrict__ out)
{
    __shared__ __align__(32) __bf16 plds[4][16][KB];  // per-wave P staging (8 KB)
    __shared__ __align__(16) float  mbuf[2][32][80];  // split-K merge (20 KB)

    const int b    = blockIdx.y;
    const int wv   = threadIdx.x >> 5;
    const int lane = threadIdx.x & 31;
    const int m    = lane & 15;     // A-row / B-col / C-col index
    const int h    = lane >> 4;     // wave half
    const int qt   = wv >> 1;       // query tile within block
    const int kh   = wv & 1;        // key half handled by this wave
    const int qbase  = (blockIdx.x * 2 + qt) * 16;
    const int kstart = kh * KHALF;

    const __bf16* Q   = Qt + (size_t)b * NN * CC;
    const __bf16* K   = Kt + (size_t)b * NN * CC;
    const __bf16* VXp = Vx + (size_t)b * CC * NN;
    const __bf16* VYp = Vy + (size_t)b * CC * NN;

    // Q fragments for this wave's 16 rows (K=64 channels -> two 16x32 frags)
    const __bf16* qrow = Q + (size_t)(qbase + m) * CC;
    const v16bf aq0 = load_a_row(qrow, h, 0);
    const v16bf aq1 = load_a_row(qrow, h, 32);

    // Accumulators: 4 channel quarters x {x,y}; C/D layout:
    // element r of a lane = (query row r+8h, channel 16*q + m).
    v8f ax[4] = {}, ay[4] = {};
    float rmax[8], ps[8];           // row-wide max, PER-LANE partial sumexp
#pragma unroll
    for (int r = 0; r < 8; ++r) { rmax[r] = -3.0e38f; ps[r] = 0.0f; }

    for (int it = 0; it < KHALF; it += KB) {
        const int kb = kstart + it;

        // ---- scores S(16q x 64k), contraction over 64 channels ----
        v8f ss[4] = {};
#pragma unroll
        for (int sub = 0; sub < 4; ++sub) {
            const __bf16* kp = K + (size_t)(kb + sub * 16 + m) * CC + h * 16;
            v16bf bk0 = *(const v16bf*)(kp);        // channels 0..31 half
            v16bf bk1 = *(const v16bf*)(kp + 32);   // channels 32..63 half
            ss[sub] = wmma_bf16(aq0, bk0, ss[sub]);
            ss[sub] = wmma_bf16(aq1, bk1, ss[sub]);
        }

        // ---- online softmax: row max via DPP butterflies, deferred sums ----
        float alpha[8];
#pragma unroll
        for (int r = 0; r < 8; ++r) {
            float bm = fmaxf(fmaxf(ss[0][r], ss[1][r]),
                             fmaxf(ss[2][r], ss[3][r]));
            bm = rowmax16(bm);
            float nm = fmaxf(rmax[r], bm);
            alpha[r] = __expf(rmax[r] - nm);
            float p0 = __expf(ss[0][r] - nm);
            float p1 = __expf(ss[1][r] - nm);
            float p2 = __expf(ss[2][r] - nm);
            float p3 = __expf(ss[3][r] - nm);
            ps[r] = alpha[r] * ps[r] + ((p0 + p1) + (p2 + p3));
            rmax[r] = nm;
            __bf16* prow = &plds[wv][r + 8 * h][m];   // row=query, col=key
            prow[0]  = f2bf(p0);
            prow[16] = f2bf(p1);
            prow[32] = f2bf(p2);
            prow[48] = f2bf(p3);
        }
#pragma unroll
        for (int r = 0; r < 8; ++r) {
            const float a = alpha[r];
            ax[0][r] *= a; ax[1][r] *= a; ax[2][r] *= a; ax[3][r] *= a;
            ay[0][r] *= a; ay[1][r] *= a; ay[2][r] *= a; ay[3][r] *= a;
        }

        // Wave-local LDS drain: P tile is private to this wave and DS ops are
        // in-order within a wave, so a dscnt wait (not a WG barrier) suffices.
        asm volatile("s_wait_dscnt 0x0" ::: "memory");

        // P as two 16x32 A fragments (keys kb..+31 and kb+32..+63)
        const __bf16* prow = &plds[wv][m][0];
        v16bf ap0 = cat16(*(const v8bf*)(prow + h * 16),
                          *(const v8bf*)(prow + 8 + h * 16));
        v16bf ap1 = cat16(*(const v8bf*)(prow + 32 + h * 16),
                          *(const v8bf*)(prow + 40 + h * 16));

        // ---- O += P * V  (B frag: col = channel, 16 consecutive keys) ----
#pragma unroll
        for (int q = 0; q < 4; ++q) {
            const __bf16* vx = VXp + (size_t)(q * 16 + m) * NN + kb + h * 16;
            const __bf16* vy = VYp + (size_t)(q * 16 + m) * NN + kb + h * 16;
            ax[q] = wmma_bf16(ap0, *(const v16bf*)(vx),      ax[q]);
            ax[q] = wmma_bf16(ap1, *(const v16bf*)(vx + 32), ax[q]);
            ay[q] = wmma_bf16(ap0, *(const v16bf*)(vy),      ay[q]);
            ay[q] = wmma_bf16(ap1, *(const v16bf*)(vy + 32), ay[q]);
        }
    }

    // ---- split-K merge + epilogue ----
    __syncthreads();
    if (kh == 1) {
        float* mb = &mbuf[qt][lane][0];
#pragma unroll
        for (int q = 0; q < 4; ++q)
#pragma unroll
            for (int r = 0; r < 8; ++r) {
                mb[q * 8 + r]      = ax[q][r];
                mb[32 + q * 8 + r] = ay[q][r];
            }
#pragma unroll
        for (int r = 0; r < 8; ++r) { mb[64 + r] = rmax[r]; mb[72 + r] = ps[r]; }
    }
    __syncthreads();
    if (kh == 0) {
        const float* mb = &mbuf[qt][lane][0];
        const float g  = gptr[0];
        const float bt = bptr[0];
        float* outx = out + (size_t)b * CC * NN;
        float* outy = out + (size_t)NB * CC * NN + (size_t)b * CC * NN;
#pragma unroll
        for (int r = 0; r < 8; ++r) {
            const float m2  = mb[64 + r];
            const float ps2 = mb[72 + r];
            const float M  = fmaxf(rmax[r], m2);
            const float a1 = __expf(rmax[r] - M);
            const float a2 = __expf(m2 - M);
            const float rowsum = rowsum16(a1 * ps[r] + a2 * ps2);
            const float sg = g / rowsum;
            const float sb = bt / rowsum;
            const int   qq = qbase + r + 8 * h;
#pragma unroll
            for (int q = 0; q < 4; ++q) {
                const float vx = a1 * ax[q][r] + a2 * mb[q * 8 + r];
                const float vy = a1 * ay[q][r] + a2 * mb[32 + q * 8 + r];
                outx[(size_t)(q * 16 + m) * NN + qq] = sg * vx;
                outy[(size_t)(q * 16 + m) * NN + qq] = sb * vy;
            }
        }
    }
}

extern "C" void kernel_launch(void* const* d_in, const int* in_sizes, int n_in,
                              void* d_out, int out_size, void* d_ws, size_t ws_size,
                              hipStream_t stream) {
    (void)in_sizes; (void)n_in; (void)out_size; (void)ws_size;
    const float* x     = (const float*)d_in[0];
    const float* y     = (const float*)d_in[1];
    const float* Wq    = (const float*)d_in[2];
    const float* bq    = (const float*)d_in[3];
    const float* Wk    = (const float*)d_in[4];
    const float* bk    = (const float*)d_in[5];
    const float* Wvx   = (const float*)d_in[6];
    const float* bvx   = (const float*)d_in[7];
    const float* Wvy   = (const float*)d_in[8];
    const float* bvy   = (const float*)d_in[9];
    const float* gamma = (const float*)d_in[10];
    const float* beta  = (const float*)d_in[11];

    // Workspace: 4 bf16 matrices of B*N*C elements (9.4 MB total).
    const size_t mat = (size_t)NB * NN * CC;
    __bf16* Qt = (__bf16*)d_ws;
    __bf16* Kt = Qt + mat;
    __bf16* Vx = Kt + mat;
    __bf16* Vy = Vx + mat;

    dim3 g1(NN / 4, NB, 4);   // 4 pixels x 64 outs per 256-thread block
    proj_kernel<<<g1, 256, 0, stream>>>(x, y, Wq, bq, Wk, bk,
                                        Wvx, bvx, Wvy, bvy,
                                        Qt, Kt, Vx, Vy);

    dim3 g2(NN / 32, NB, 1);  // 2 query tiles x 2 key-halves per 128-thread block
    attn_kernel<<<g2, 128, 0, stream>>>(Qt, Kt, Vx, Vy, gamma, beta,
                                        (float*)d_out);
}
